// CharCPRNN_23545010717004
// MI455X (gfx1250) — matrix operand order repaired and checked by reading, not compiled, not run on hardware
//
#include <hip/hip_runtime.h>
#include <math.h>

// Problem dims
#define VOCABx 128
#define Ex     512
#define Hx     1024
#define Rx     512
#define Bx     64
#define Lx     512

#define NBLK     32     // persistent blocks for the scan
#define NTHREADS 256    // 8 waves of 32

// dynamic LDS: Vw slice 32x1024 + Fo slice 32x512 + FhT slice 16x1024 (bf16)
#define SMEM_ELTS  (32 * Hx + 32 * Rx + 16 * Hx)   // 65536 ushorts
#define SMEM_BYTES (SMEM_ELTS * 2)                 // 128 KB

typedef __bf16        v16bf __attribute__((ext_vector_type(16)));
typedef float         v8f   __attribute__((ext_vector_type(8)));
typedef unsigned int  u32x4 __attribute__((ext_vector_type(4)));

union Frag { v16bf bf; u32x4 u[2]; };

__device__ __forceinline__ unsigned short f2bf(float f) {
  unsigned u = __float_as_uint(f);
  unsigned r = ((u >> 16) & 1u) + 0x7fffu;   // round-to-nearest-even
  return (unsigned short)((u + r) >> 16);
}

__device__ __forceinline__ v8f wmma_bf16(Frag a, Frag b, v8f c) {
  // (neg_a, A, neg_b, B, c_mod, C, reuse_a, reuse_b)
  return __builtin_amdgcn_wmma_f32_16x16x32_bf16(false, a.bf, false, b.bf,
                                                 (short)0, c, false, false);
}

// A fragment: row-major [M][K] bf16. lane m holds row m; lane half selects
// K {k0+0..7, k0+16..23} (half 0) or {k0+8..15, k0+24..31} (half 1).
__device__ __forceinline__ Frag load_a(const unsigned short* row, int k0, int hf) {
  Frag f;
  f.u[0] = *(const u32x4*)(row + k0 + hf * 8);
  f.u[1] = *(const u32x4*)(row + k0 + 16 + hf * 8);
  return f;
}
// B fragment from BT stored row-major [N][K] (B[k][n] = BT[n][k]).
// lane n holds column n; half 0 -> K {k0+0..7, k0+16..23}, half 1 -> K {k0+8..15, k0+24..31}.
__device__ __forceinline__ Frag load_bt(const unsigned short* row, int k0, int hf) {
  Frag f;
  f.u[0] = *(const u32x4*)(row + k0 + hf * 8);
  f.u[1] = *(const u32x4*)(row + k0 + 16 + hf * 8);
  return f;
}

// coalesced 16-byte block copy global -> LDS (n multiple of 8)
__device__ __forceinline__ void copy_smem(unsigned short* dst,
                                          const unsigned short* src,
                                          int n, int tid) {
  u32x4* d = (u32x4*)dst;
  const u32x4* s = (const u32x4*)src;
  for (int i = tid; i < (n >> 3); i += NTHREADS) d[i] = s[i];
}

// --------------------------------------------------------------------------
// Device-wide barrier (persistent-kernel sync). Release fence flushes this
// wave's writes to device scope, acquire fence invalidates stale lines.
__device__ __forceinline__ void sync_grid(unsigned* cnt, unsigned target, int tid) {
  __builtin_amdgcn_fence(__ATOMIC_RELEASE, "agent");
  __syncthreads();
  if (tid == 0) {
    __hip_atomic_fetch_add(cnt, 1u, __ATOMIC_RELEASE, __HIP_MEMORY_SCOPE_AGENT);
    while (__hip_atomic_load(cnt, __ATOMIC_ACQUIRE, __HIP_MEMORY_SCOPE_AGENT) < target)
      __builtin_amdgcn_s_sleep(2);
  }
  __syncthreads();
  __builtin_amdgcn_fence(__ATOMIC_ACQUIRE, "agent");
}

// --------------------------------------------------------------------------
// Setup kernels
__global__ void k_cvt(const float* __restrict__ s, unsigned short* __restrict__ d, int n) {
  for (int i = blockIdx.x * blockDim.x + threadIdx.x; i < n; i += gridDim.x * blockDim.x)
    d[i] = f2bf(s[i]);
}

// FhT[r][h] = Fh[h][r]  (so B[k][n]=Fh[k][n]=FhT[n][k] is contiguous per lane)
__global__ void k_fht(const float* __restrict__ Fh, unsigned short* __restrict__ FhT) {
  int i = blockIdx.x * blockDim.x + threadIdx.x;
  if (i < Rx * Hx) {
    int r = i / Hx, hh = i % Hx;
    FhT[i] = f2bf(Fh[hh * Rx + r]);
  }
}

// EFx[v][r] = (emb @ Fx)[v][r];  EU[v][h] = (emb @ U_w^T)[v][h] + bias[h]
__global__ void k_tables(const float* __restrict__ emb, const float* __restrict__ Fx,
                         const float* __restrict__ Uw, const float* __restrict__ bias,
                         float* __restrict__ EFx, float* __restrict__ EU) {
  int i = blockIdx.x * blockDim.x + threadIdx.x;
  if (i < VOCABx * Rx) {
    int v = i / Rx, r = i % Rx;
    float s = 0.f;
    for (int e = 0; e < Ex; ++e) s += emb[v * Ex + e] * Fx[e * Rx + r];
    EFx[i] = s;
  } else if (i < VOCABx * Rx + VOCABx * Hx) {
    int j = i - VOCABx * Rx;
    int v = j / Hx, hh = j % Hx;
    float s = bias[hh];
    for (int e = 0; e < Ex; ++e) s += emb[v * Ex + e] * Uw[hh * Ex + e];
    EU[j] = s;
  }
}

// h0 (bf16) from input h; zero the barrier counter (fresh every launch)
__global__ void k_init(const float* __restrict__ h, unsigned short* __restrict__ h0,
                       unsigned* __restrict__ cnt) {
  int i = blockIdx.x * blockDim.x + threadIdx.x;
  if (i < Bx * Hx) h0[i] = f2bf(h[i]);
  if (i == 0) *cnt = 0u;
}

// --------------------------------------------------------------------------
// Persistent scan: 32 blocks, 2 device barriers per step, balanced waves.
// Weight slices (invariant over t) are staged once into LDS: all B fragments
// are LDS-served for all 512 steps; only h/wbuf stream from global.
//  Phase 1 (all 8 waves, 48 WMMAs each):
//    - accL(tile mt,nt) = h @ V_w^T   over full K=1024             (32 WMMAs)
//    - u-partial(tile mt, K-half nt) = h @ Fh                      (16 WMMAs)
//      partials reduced across wave pairs through LDS, then
//      waves 0-3: w = u * EFx[x[:,t]] -> wbuf (bf16)
//  Phase 2 (all 8 waves, 16 WMMAs each):
//    - accL += w @ Fo^T ; h_new = tanh(accL + EU[x[:,t]])
__global__ void scan_kernel(const int* __restrict__ x,
                            const unsigned short* __restrict__ FhT,
                            const unsigned short* __restrict__ FoB,   // Fo bf16 [H][R]
                            const unsigned short* __restrict__ VwB,   // V_w bf16 [H][H]
                            const float* __restrict__ EFx,
                            const float* __restrict__ EU,
                            const unsigned short* __restrict__ h0,
                            unsigned short* __restrict__ wbuf,        // [B][R] bf16
                            unsigned short* __restrict__ hidden,      // [B][L][H] bf16
                            unsigned* __restrict__ cnt,
                            float* __restrict__ hfinal) {
  const int blk  = blockIdx.x;
  const int tid  = threadIdx.x;
  const int wave = tid >> 5;
  const int lane = tid & 31;
  const int hf   = lane >> 4;
  const int l16  = lane & 15;
  const int mt   = wave & 3;          // M-tile (batch rows 16*mt..)
  const int nt   = wave >> 2;         // N-tile for lin / K-half for u
  unsigned bar = 0;

  __shared__ float upart[8][16][17];  // padded: conflict-free half-lane access
  extern __shared__ char smem[];
  unsigned short* sVw = (unsigned short*)smem;   // [32][1024]
  unsigned short* sFo = sVw + 32 * Hx;           // [32][512]
  unsigned short* sFh = sFo + 32 * Rx;           // [16][1024]

  // stage this block's weight slices (contiguous row blocks) into LDS once
  copy_smem(sVw, VwB + (size_t)blk * 32 * Hx, 32 * Hx, tid);
  copy_smem(sFo, FoB + (size_t)blk * 32 * Rx, 32 * Rx, tid);
  copy_smem(sFh, FhT + (size_t)blk * 16 * Hx, 16 * Hx, tid);
  __syncthreads();

  const unsigned short* bVw = sVw + (size_t)(nt * 16 + l16) * Hx;  // lin cols
  const unsigned short* bFo = sFo + (size_t)(nt * 16 + l16) * Rx;  // bil cols
  const unsigned short* bFh = sFh + (size_t)l16 * Hx;              // u cols

#pragma unroll 1
  for (int t = 0; t < Lx; ++t) {
    const unsigned short* hA; size_t hstr;
    if (t == 0) { hA = h0; hstr = (size_t)Hx; }
    else        { hA = hidden + (size_t)(t - 1) * Hx; hstr = (size_t)Lx * Hx; }

    const unsigned short* harow = hA + (size_t)(mt * 16 + l16) * hstr;
    const int ncol = blk * 32 + nt * 16 + l16;

    // ---- Phase 1a: lin tile, full K (B from LDS) ----
    v8f accL = {0.f, 0.f, 0.f, 0.f, 0.f, 0.f, 0.f, 0.f};
#pragma unroll 4
    for (int k0 = 0; k0 < Hx; k0 += 32) {
      __builtin_prefetch(harow + k0 + 256, 0, 1);   // stream h rows
      accL = wmma_bf16(load_a(harow, k0, hf), load_bt(bVw, k0, hf), accL);
    }
    // ---- Phase 1b: u partial, K-half selected by nt (B from LDS) ----
    {
      v8f accU = {0.f, 0.f, 0.f, 0.f, 0.f, 0.f, 0.f, 0.f};
      const int kbeg = nt * (Hx / 2), kend = kbeg + (Hx / 2);
#pragma unroll 4
      for (int k0 = kbeg; k0 < kend; k0 += 32)
        accU = wmma_bf16(load_a(harow, k0, hf), load_bt(bFh, k0, hf), accU);
#pragma unroll
      for (int j = 0; j < 8; ++j)
        upart[wave][hf * 8 + j][l16] = accU[j];      // ds_store
    }
    __syncthreads();
    if (wave < 4) {                                  // wave-uniform branch
      const int r = blk * 16 + l16;
#pragma unroll
      for (int j = 0; j < 8; ++j) {
        int   ml = hf * 8 + j;
        int   mb = mt * 16 + ml;                     // batch row
        float uv = upart[mt][ml][l16] + upart[mt + 4][ml][l16];   // ds_load x2
        int   xv = x[mb * Lx + t];
        wbuf[(size_t)mb * Rx + r] = f2bf(uv * EFx[xv * Rx + r]);
      }
    }
    ++bar; sync_grid(cnt, NBLK * bar, tid);

    // ---- Phase 2: bilinear part accumulates into live lin registers ----
    {
      const unsigned short* arow = wbuf + (size_t)(mt * 16 + l16) * Rx;
#pragma unroll 4
      for (int k0 = 0; k0 < Rx; k0 += 32)
        accL = wmma_bf16(load_a(arow, k0, hf), load_bt(bFo, k0, hf), accL);
#pragma unroll
      for (int j = 0; j < 8; ++j) {
        int   mb = mt * 16 + hf * 8 + j;
        int   xv = x[mb * Lx + t];
        float hv = tanhf(accL[j] + EU[(size_t)xv * Hx + ncol]);
        hidden[(size_t)mb * (size_t)(Lx * Hx) + (size_t)t * Hx + ncol] = f2bf(hv);
        if (t == Lx - 1) hfinal[mb * Hx + ncol] = hv;   // exact fp32 final state
      }
    }
    ++bar; sync_grid(cnt, NBLK * bar, tid);
  }
}

// --------------------------------------------------------------------------
// Decode: logits[m][v] = hidden[m][:] . dec_w[v][:] + dec_b[v]
// grid = (M/64, VOCAB/32); block 256 = 8 waves = 4 Mtiles x 2 Ntiles.
__global__ void decode_kernel(const unsigned short* __restrict__ hidden,
                              const unsigned short* __restrict__ decw,  // bf16 [V][H]
                              const float* __restrict__ decb,
                              float* __restrict__ logits) {
  const int tid  = threadIdx.x;
  const int wave = tid >> 5, lane = tid & 31, hf = lane >> 4, l16 = lane & 15;
  const int mt = wave & 3, nt = wave >> 2;
  const int mrow = blockIdx.x * 64 + mt * 16 + l16;
  const int ncol = blockIdx.y * 32 + nt * 16 + l16;
  v8f acc = {0.f, 0.f, 0.f, 0.f, 0.f, 0.f, 0.f, 0.f};
  const unsigned short* arow = hidden + (size_t)mrow * Hx;
  const unsigned short* brow = decw + (size_t)ncol * Hx;
#pragma unroll 4
  for (int k0 = 0; k0 < Hx; k0 += 32)
    acc = wmma_bf16(load_a(arow, k0, hf), load_bt(brow, k0, hf), acc);
#pragma unroll
  for (int j = 0; j < 8; ++j) {
    int m = blockIdx.x * 64 + mt * 16 + hf * 8 + j;
    logits[(size_t)m * VOCABx + ncol] = acc[j] + decb[ncol];
  }
}

// --------------------------------------------------------------------------
extern "C" void kernel_launch(void* const* d_in, const int* in_sizes, int n_in,
                              void* d_out, int out_size, void* d_ws, size_t ws_size,
                              hipStream_t stream) {
  (void)in_sizes; (void)n_in; (void)out_size; (void)ws_size;
  const int*   x    = (const int*)d_in[0];
  const float* h    = (const float*)d_in[1];
  const float* emb  = (const float*)d_in[2];
  const float* Fh   = (const float*)d_in[3];
  const float* Fx   = (const float*)d_in[4];
  const float* Fo   = (const float*)d_in[5];
  const float* Uw   = (const float*)d_in[6];
  const float* Vw   = (const float*)d_in[7];
  const float* bias = (const float*)d_in[8];
  const float* decw = (const float*)d_in[9];
  const float* decb = (const float*)d_in[10];

  char* ws = (char*)d_ws;
  size_t o = 0;
  unsigned short* FhT_b = (unsigned short*)(ws + o); o += (size_t)Rx * Hx * 2;     // 1 MB
  unsigned short* Fo_b  = (unsigned short*)(ws + o); o += (size_t)Hx * Rx * 2;     // 1 MB
  unsigned short* Vw_b  = (unsigned short*)(ws + o); o += (size_t)Hx * Hx * 2;     // 2 MB
  unsigned short* dec_b16 = (unsigned short*)(ws + o); o += (size_t)VOCABx * Hx * 2;
  float*          EFx_t = (float*)(ws + o); o += (size_t)VOCABx * Rx * 4;
  float*          EU_t  = (float*)(ws + o); o += (size_t)VOCABx * Hx * 4;
  unsigned short* h0_b  = (unsigned short*)(ws + o); o += (size_t)Bx * Hx * 2;
  unsigned short* wbuf  = (unsigned short*)(ws + o); o += (size_t)Bx * Rx * 2;
  unsigned*       cnt   = (unsigned*)(ws + o);       o += 256;
  unsigned short* hidden = (unsigned short*)(ws + o); o += (size_t)Bx * Lx * Hx * 2; // 64 MB

  float* logits = (float*)d_out;
  float* hfinal = logits + (size_t)Bx * Lx * VOCABx;

  // allow >64KB dynamic LDS for the scan kernel (no-op if already allowed)
  (void)hipFuncSetAttribute((const void*)scan_kernel,
                            hipFuncAttributeMaxDynamicSharedMemorySize, SMEM_BYTES);

  // weight conversions + tables + init
  k_cvt<<<2048, NTHREADS, 0, stream>>>(Fo,   Fo_b,  Hx * Rx);
  k_cvt<<<4096, NTHREADS, 0, stream>>>(Vw,   Vw_b,  Hx * Hx);
  k_cvt<<<512,  NTHREADS, 0, stream>>>(decw, dec_b16, VOCABx * Hx);
  k_fht<<<2048, NTHREADS, 0, stream>>>(Fh, FhT_b);
  k_tables<<<768, NTHREADS, 0, stream>>>(emb, Fx, Uw, bias, EFx_t, EU_t);
  k_init<<<256, NTHREADS, 0, stream>>>(h, h0_b, cnt);

  // persistent recurrent scan (weights LDS-resident for all 512 steps)
  scan_kernel<<<NBLK, NTHREADS, SMEM_BYTES, stream>>>(x, FhT_b, Fo_b, Vw_b,
                                                      EFx_t, EU_t, h0_b, wbuf,
                                                      hidden, cnt, hfinal);

  // decode all timesteps
  decode_kernel<<<dim3((Bx * Lx) / 64, VOCABx / 32), NTHREADS, 0, stream>>>(
      hidden, dec_b16, decb, logits);
}
